// TempPRGCN_44418551775494
// MI455X (gfx1250) — matrix operand order repaired and checked by reading, not compiled, not run on hardware
//
#include <hip/hip_runtime.h>

#define T_ 128
#define K_ 14
#define F_ 1024
#define TK_ 1792            // T_*K_
#define TKF_ ((size_t)TK_ * F_)

typedef __attribute__((ext_vector_type(16))) _Float16     v16h;
typedef __attribute__((ext_vector_type(8)))  float        v8f;
typedef __attribute__((ext_vector_type(4)))  unsigned int v4u;
typedef __attribute__((ext_vector_type(8)))  unsigned int v8u;

union FragU { v8u u; v16h h; };

__device__ __forceinline__ int imin(int a, int b) { return a < b ? a : b; }

#define WMMA_F16(A, B, Cacc) \
  __builtin_amdgcn_wmma_f32_16x16x32_f16(false, (A), false, (B), (short)0, (Cacc), false, false)

// ---- fragment loaders -------------------------------------------------------
// A (16x32 f16) from a row-major f16 matrix. Layout: lane L (r=L&15,
// half=L>>4): V0-7 hold K = 16*half..16*half+15 (matches packed B).
// Both chunks are 16 contiguous bytes -> two b128 loads.
__device__ __forceinline__ v16h load_a_frag(const _Float16* Arow, int kbase, int half) {
  FragU f;
  f.u.lo = *(const v4u*)(Arow + kbase + 16 * half);
  f.u.hi = *(const v4u*)(Arow + kbase + 16 * half + 8);
  return f.h;
}
// B (32x16 f16) from fragment-linear packed panel: tile holds 32 lanes * 16 f16,
// lane n (col=n&15, khalf=n>>4) holds K = 16*khalf+0..15 consecutively.
__device__ __forceinline__ v16h load_b_frag(const _Float16* tile, int lane) {
  FragU f;
  const _Float16* p = tile + lane * 16;
  f.u.lo = *(const v4u*)(p);
  f.u.hi = *(const v4u*)(p + 8);
  return f.h;
}

// ---- GCN-normalized adjacency ----------------------------------------------
__global__ void k_an(const float* __restrict__ A, float* __restrict__ An) {
  __shared__ float dinv[K_];
  int tid = threadIdx.x;
  if (tid < K_) {
    float s = 1.0f;  // self loop
    for (int j = 0; j < K_; ++j) s += (A[tid * K_ + j] != 0.0f) ? 1.0f : 0.0f;
    dinv[tid] = rsqrtf(s);
  }
  __syncthreads();
  if (tid < K_ * K_) {
    int k = tid / K_, j = tid % K_;
    float ah = ((A[tid] != 0.0f) ? 1.0f : 0.0f) + ((k == j) ? 1.0f : 0.0f);
    An[tid] = dinv[k] * ah * dinv[j];
  }
}

// ---- pack a 1024x1024 row-major f32 weight into fragment-linear f16 ---------
__global__ __launch_bounds__(256) void k_pack(const float* __restrict__ W,
                                              _Float16* __restrict__ P) {
  int idx = blockIdx.x * blockDim.x + threadIdx.x;  // 0 .. 2^20-1
  int e    = idx & 15;
  int lane = (idx >> 4) & 31;
  int kt   = (idx >> 9) & 31;
  int nt   = idx >> 14;
  int k = kt * 32 + 16 * (lane >> 4) + e;
  int n = nt * 16 + (lane & 15);
  P[idx] = (_Float16)W[(size_t)k * F_ + n];
}

// ---- bilinear 64 -> 32 downsample (both dims), emit f32 + f16 ---------------
__global__ __launch_bounds__(256) void k_down(const float* __restrict__ x,
                                              float* __restrict__ ff,
                                              _Float16* __restrict__ fh) {
  int idx = blockIdx.x * blockDim.x + threadIdx.x;  // TK_*1024
  int p = idx & 31, o = (idx >> 5) & 31, tk = idx >> 10;
  float co = o * (63.0f / 31.0f);
  int lo = (int)floorf(co); int ho = imin(lo + 1, 63); float wo = co - (float)lo;
  float cp = p * (63.0f / 31.0f);
  int lp = (int)floorf(cp); int hp = imin(lp + 1, 63); float wp = cp - (float)lp;
  const float* xb = x + (size_t)tk * 4096;
  float v = (1.f - wo) * (1.f - wp) * xb[lo * 64 + lp] +
            (1.f - wo) * wp         * xb[lo * 64 + hp] +
            wo         * (1.f - wp) * xb[ho * 64 + lp] +
            wo         * wp         * xb[ho * 64 + hp];
  ff[idx] = v;
  fh[idx] = (_Float16)v;
}

// ---- WMMA GEMM: C[M,N](f32) = A[M,K](f16 rowmajor) x Bpacked + optional bias
// grid (N/512, M/32), block 256 = 8 waves; each wave owns a 32x64 block
// (2 M-tiles x 4 N-tiles, 8 accumulators). Ping-pong software pipeline: two
// persistent fragment sets; while set0's WMMAs execute, set1's 12 b128 loads
// are in flight (and vice versa) -- no register rotation copies.
__global__ __launch_bounds__(256) void k_gemm(const _Float16* __restrict__ Ah,
                                              const _Float16* __restrict__ Bp,
                                              float* __restrict__ C,
                                              const float* __restrict__ bias,
                                              int M, int N, int K) {
  const int lane = threadIdx.x & 31;
  const int wv   = threadIdx.x >> 5;
  const int m0   = blockIdx.y << 5;            // 32 rows
  const int nt0  = (blockIdx.x * 8 + wv) * 4;  // 4 n-tiles of 16
  const int KT   = K >> 5;
  const int r = lane & 15, half = lane >> 4;
  const _Float16* Arow0 = Ah + (size_t)(m0 + r) * K;
  const _Float16* Arow1 = Arow0 + (size_t)16 * K;
  v8f acc[2][4] = {};
  v16h a0_0, a1_0, a0_1, a1_1;
  v16h b0[4], b1[4];

#define GLOADK(A0, A1, B, kt_)                                                 \
  do {                                                                         \
    A0 = load_a_frag(Arow0, (kt_) << 5, half);                                 \
    A1 = load_a_frag(Arow1, (kt_) << 5, half);                                 \
    _Pragma("unroll")                                                          \
    for (int j = 0; j < 4; ++j)                                                \
      B[j] = load_b_frag(Bp + (((size_t)(nt0 + j) * KT + (kt_)) << 9), lane);  \
  } while (0)

#define GCONSUME(A0, A1, B)                                                    \
  do {                                                                         \
    _Pragma("unroll")                                                          \
    for (int j = 0; j < 4; ++j) {                                              \
      acc[0][j] = WMMA_F16(A0, B[j], acc[0][j]);                               \
      acc[1][j] = WMMA_F16(A1, B[j], acc[1][j]);                               \
    }                                                                          \
  } while (0)

  GLOADK(a0_0, a1_0, b0, 0);
  GLOADK(a0_1, a1_1, b1, 1);
  for (int kt = 0; kt < KT - 2; kt += 2) {
    GCONSUME(a0_0, a1_0, b0);
    GLOADK(a0_0, a1_0, b0, kt + 2);
    GCONSUME(a0_1, a1_1, b1);
    GLOADK(a0_1, a1_1, b1, kt + 3);
  }
  GCONSUME(a0_0, a1_0, b0);
  GCONSUME(a0_1, a1_1, b1);
#undef GLOADK
#undef GCONSUME

#pragma unroll
  for (int mi = 0; mi < 2; ++mi) {
#pragma unroll
    for (int j = 0; j < 4; ++j) {
      int col = ((nt0 + j) << 4) + r;
      float bv = bias ? bias[col] : 0.0f;
#pragma unroll
      for (int e = 0; e < 8; ++e) {
        int row = m0 + (mi << 4) + e + (half << 3);
        C[(size_t)row * N + col] = acc[mi][j][e] + bv;
      }
    }
  }
}

// ---- An-mix over nodes + bias (+relu), emit f32/f16 -------------------------
__global__ __launch_bounds__(256) void k_mix(const float* __restrict__ An,
                                             const float* __restrict__ Y,
                                             const float* __restrict__ bias,
                                             float* __restrict__ outf,
                                             _Float16* __restrict__ outh,
                                             int relu) {
  int idx = blockIdx.x * blockDim.x + threadIdx.x;  // (t*K+k)*F + f
  int f = idx & (F_ - 1);
  int tk = idx >> 10;
  int k = tk % K_, t = tk / K_;
  const float* Yt = Y + (size_t)t * K_ * F_ + f;
  float s = bias[f];
#pragma unroll
  for (int j = 0; j < K_; ++j) s += An[k * K_ + j] * Yt[(size_t)j * F_];
  if (relu) s = fmaxf(s, 0.0f);
  if (outf) outf[idx] = s;
  if (outh) outh[idx] = (_Float16)s;
}

// ---- one 16x16 output tile of (H[16x1024] @ W), H in LDS, ping-pong pipeline
__device__ __forceinline__ v8f tile_gemm_lds(const _Float16* Asrc,
                                             const _Float16* __restrict__ Bp,
                                             int nt, int lane) {
  const int r = lane & 15, half = lane >> 4;
  const _Float16* Arow = Asrc + r * F_;
  const _Float16* Bt = Bp + ((size_t)(nt << 5) << 9);
  v8f acc = {};
  v16h af0 = load_a_frag(Arow, 0, half);
  v16h bf0 = load_b_frag(Bt, lane);
  v16h af1 = load_a_frag(Arow, 1 << 5, half);
  v16h bf1 = load_b_frag(Bt + ((size_t)1 << 9), lane);
  for (int kt = 0; kt < 30; kt += 2) {
    acc = WMMA_F16(af0, bf0, acc);
    af0 = load_a_frag(Arow, (kt + 2) << 5, half);
    bf0 = load_b_frag(Bt + ((size_t)(kt + 2) << 9), lane);
    acc = WMMA_F16(af1, bf1, acc);
    af1 = load_a_frag(Arow, (kt + 3) << 5, half);
    bf1 = load_b_frag(Bt + ((size_t)(kt + 3) << 9), lane);
  }
  acc = WMMA_F16(af0, bf0, acc);
  acc = WMMA_F16(af1, bf1, acc);
  return acc;
}

// ---- sequential GRU-style scan; blockIdx.x = direction (0 fwd, 1 bwd) -------
__global__ __launch_bounds__(1024) void k_scan(const float* __restrict__ Ub,
                                               const _Float16* __restrict__ Wb,
                                               const int* __restrict__ vid,
                                               float* __restrict__ scratch,
                                               float* __restrict__ Hout) {
  const int dir = blockIdx.x;
  const int tid = threadIdx.x, lane = tid & 31, wv = tid >> 5;
  const float* U0 = Ub + ((size_t)dir * 3 + 0) * TKF_;
  const float* U1 = Ub + ((size_t)dir * 3 + 1) * TKF_;
  const float* U2 = Ub + ((size_t)dir * 3 + 2) * TKF_;
  const _Float16* W0 = Wb + ((size_t)dir * 3 + 0) * (1u << 20);
  const _Float16* W1 = Wb + ((size_t)dir * 3 + 1) * (1u << 20);
  const _Float16* W2 = Wb + ((size_t)dir * 3 + 2) * (1u << 20);
  float* Z  = scratch + (size_t)dir * 2 * 16384;
  float* Hf = Z + 16384;
  float* Ho = Hout + (size_t)dir * TKF_;

  __shared__ __align__(16) _Float16 Hh[16 * F_];   // H, rows 14..15 stay zero
  __shared__ __align__(16) _Float16 HRh[16 * F_];  // H*R

  for (int i = tid; i < 16 * F_; i += 1024) {
    Hh[i] = (_Float16)0.0f; HRh[i] = (_Float16)0.0f; Hf[i] = 0.0f;
  }
  __threadfence(); __syncthreads();

  for (int s = 0; s < T_; ++s) {
    const int t = dir ? (T_ - 1 - s) : s;
    bool reset = (s == 0);
    if (!reset) { int tp = dir ? t + 1 : t - 1; reset = (vid[t] != vid[tp]); }
    if (reset && s) {
      for (int i = tid; i < 16 * F_; i += 1024) { Hh[i] = (_Float16)0.0f; Hf[i] = 0.0f; }
    }
    __threadfence(); __syncthreads();

    // Stage A: gates Z (g=0) and R (g=1): 2 gates x 64 n-tiles = 128 jobs
#pragma unroll
    for (int jj = 0; jj < 4; ++jj) {
      int job = wv + (jj << 5);
      int g = job >> 6, nt = job & 63;
      v8f acc = tile_gemm_lds(Hh, g ? W1 : W0, nt, lane);
      const float* Ug = g ? U1 : U0;
      int r = lane & 15, half = lane >> 4, col = (nt << 4) + r;
#pragma unroll
      for (int e = 0; e < 8; ++e) {
        int row = e + (half << 3);
        if (row < K_) {
          float pre = acc[e] + Ug[((size_t)t * K_ + row) * F_ + col];
          float sg = 1.0f / (1.0f + expf(-pre));
          if (g == 0) Z[row * F_ + col] = sg;
          else        HRh[row * F_ + col] = (_Float16)(Hf[row * F_ + col] * sg);
        }
      }
    }
    __threadfence(); __syncthreads();

    // Stage B: candidate + combine: 64 n-tile jobs
#pragma unroll
    for (int jj = 0; jj < 2; ++jj) {
      int nt = wv + (jj << 5);
      v8f acc = tile_gemm_lds(HRh, W2, nt, lane);
      int r = lane & 15, half = lane >> 4, col = (nt << 4) + r;
#pragma unroll
      for (int e = 0; e < 8; ++e) {
        int row = e + (half << 3);
        if (row < K_) {
          float ht = tanhf(acc[e] + U2[((size_t)t * K_ + row) * F_ + col]);
          float z = Z[row * F_ + col];
          float h = Hf[row * F_ + col];
          float hn = z * h + (1.0f - z) * ht;
          Hf[row * F_ + col] = hn;
          Hh[row * F_ + col] = (_Float16)hn;
          Ho[((size_t)t * K_ + row) * F_ + col] = hn;
        }
      }
    }
    __threadfence(); __syncthreads();
  }
}

// ---- bilinear 32 -> 64 upsample of (Hf+Hb) + sigmoid ------------------------
__global__ __launch_bounds__(256) void k_up(const float* __restrict__ Hdir,
                                            float* __restrict__ out) {
  int idx = blockIdx.x * blockDim.x + threadIdx.x;  // TK_*4096
  int j = idx & 63, i = (idx >> 6) & 63, tk = idx >> 12;
  float ci = i * (31.0f / 63.0f);
  int li = (int)floorf(ci); int hi2 = imin(li + 1, 31); float wi = ci - (float)li;
  float cj = j * (31.0f / 63.0f);
  int lj = (int)floorf(cj); int hj = imin(lj + 1, 31); float wj = cj - (float)lj;
  const float* ha = Hdir + (size_t)tk * F_;          // forward
  const float* hb = Hdir + TKF_ + (size_t)tk * F_;   // backward
  int i00 = li * 32 + lj, i01 = li * 32 + hj, i10 = hi2 * 32 + lj, i11 = hi2 * 32 + hj;
  float v = (1.f - wi) * (1.f - wj) * (ha[i00] + hb[i00]) +
            (1.f - wi) * wj         * (ha[i01] + hb[i01]) +
            wi         * (1.f - wj) * (ha[i10] + hb[i10]) +
            wi         * wj         * (ha[i11] + hb[i11]);
  out[idx] = 1.0f / (1.0f + expf(-v));
}

extern "C" void kernel_launch(void* const* d_in, const int* in_sizes, int n_in,
                              void* d_out, int out_size, void* d_ws, size_t ws_size,
                              hipStream_t stream) {
  (void)in_sizes; (void)n_in; (void)out_size; (void)ws_size;
  const float* x      = (const float*)d_in[0];
  const float* A      = (const float*)d_in[1];
  const float* gcn_W  = (const float*)d_in[2];   // (2,F,F)
  const float* gcn_b  = (const float*)d_in[3];   // (2,F)
  const float* gW[2]  = { (const float*)d_in[4], (const float*)d_in[8] };   // (3,F,F)
  const float* gb[2]  = { (const float*)d_in[5], (const float*)d_in[9] };   // (3,F)
  const float* lW[2]  = { (const float*)d_in[6], (const float*)d_in[10] };  // (3,2F,F)
  const float* lb[2]  = { (const float*)d_in[7], (const float*)d_in[11] };  // (3,F)
  const int*   vid    = (const int*)d_in[12];
  float* out = (float*)d_out;

  // ---- workspace carve ----
  char* w = (char*)d_ws;
  auto alloc = [&](size_t bytes) -> char* {
    char* p = w; w += (bytes + 255) & ~(size_t)255; return p;
  };
  float*     An     = (float*)alloc(K_ * K_ * sizeof(float));
  _Float16*  feath  = (_Float16*)alloc(TKF_ * 2);
  float*     featf  = (float*)alloc(TKF_ * 4);
  _Float16*  convh  = (_Float16*)alloc(TKF_ * 2);
  float*     Cbuf   = (float*)alloc(TKF_ * 4);
  float*     Ubuf   = (float*)alloc(6 * TKF_ * 4);          // [dir][gate]
  float*     Hdir   = (float*)alloc(2 * TKF_ * 4);          // Hf | Hb
  float*     scr    = (float*)alloc(2 * 2 * 16384 * 4);     // Z,Hf per dir
  const size_t PM = (size_t)1 << 20;                        // f16 elems / packed matrix
  _Float16* Pgcn  = (_Float16*)alloc(2 * PM * 2);
  _Float16* PgW   = (_Float16*)alloc(6 * PM * 2);
  _Float16* Pltop = (_Float16*)alloc(6 * PM * 2);
  _Float16* Plbot = (_Float16*)alloc(6 * PM * 2);

  const dim3 packGrid(4096), b256(256);
  const dim3 gemmGrid(2, TK_ / 32);  // N/512 strips, M/32 blocks

  // adjacency normalization
  k_an<<<1, 256, 0, stream>>>(A, An);

  // pack all weight matrices to fragment-linear f16 panels
  for (int l = 0; l < 2; ++l)
    k_pack<<<packGrid, b256, 0, stream>>>(gcn_W + (size_t)l * F_ * F_, Pgcn + l * PM);
  for (int d = 0; d < 2; ++d)
    for (int c = 0; c < 3; ++c) {
      k_pack<<<packGrid, b256, 0, stream>>>(gW[d] + (size_t)c * F_ * F_,
                                            PgW + (d * 3 + c) * PM);
      k_pack<<<packGrid, b256, 0, stream>>>(lW[d] + (size_t)c * 2 * F_ * F_,
                                            Pltop + (d * 3 + c) * PM);
      k_pack<<<packGrid, b256, 0, stream>>>(lW[d] + (size_t)c * 2 * F_ * F_ + (size_t)F_ * F_,
                                            Plbot + (d * 3 + c) * PM);
    }

  // downsample 64x64 -> 32x32, flatten to feat
  k_down<<<dim3(TK_ * F_ / 256), b256, 0, stream>>>(x, featf, feath);

  // GCN layers
  for (int l = 0; l < 2; ++l) {
    k_gemm<<<gemmGrid, b256, 0, stream>>>(feath, Pgcn + l * PM, Cbuf, nullptr,
                                          TK_, F_, F_);
    k_mix<<<dim3(TK_ * F_ / 256), b256, 0, stream>>>(An, Cbuf, gcn_b + (size_t)l * F_,
                                                     featf, feath, 1);
  }

  // per-direction input projections: U[c] = (An∘(feat@gW[c]) + gb[c]) @ lWtop[c] + lb[c]
  for (int d = 0; d < 2; ++d)
    for (int c = 0; c < 3; ++c) {
      k_gemm<<<gemmGrid, b256, 0, stream>>>(feath, PgW + (d * 3 + c) * PM, Cbuf,
                                            nullptr, TK_, F_, F_);
      k_mix<<<dim3(TK_ * F_ / 256), b256, 0, stream>>>(An, Cbuf, gb[d] + (size_t)c * F_,
                                                       nullptr, convh, 0);
      k_gemm<<<gemmGrid, b256, 0, stream>>>(convh, Pltop + (d * 3 + c) * PM,
                                            Ubuf + (size_t)(d * 3 + c) * TKF_,
                                            lb[d] + (size_t)c * F_, TK_, F_, F_);
    }

  // sequential bi-directional GRU scan (both directions concurrently, 2 blocks)
  k_scan<<<dim3(2), dim3(1024), 0, stream>>>(Ubuf, Plbot, vid, scr, Hdir);

  // upsample (Hf + Hb) 32x32 -> 64x64 + sigmoid
  k_up<<<dim3(TK_ * 4096 / 256), b256, 0, stream>>>(Hdir, out);
}
